// AttentionWithMemory_12644383719809
// MI455X (gfx1250) — matrix operand order repaired and checked by reading, not compiled, not run on hardware
//
#include <hip/hip_runtime.h>
#include <hip/hip_bf16.h>

// ---------------------------------------------------------------------------
// Local-window MHA for MI455X (gfx1250, wave32).
// All matmuls on v_wmma_f32_16x16x32_bf16; global->LDS staging uses the CDNA5
// async-to-LDS path (ASYNCcnt) when the toolchain exposes it.
// ---------------------------------------------------------------------------

typedef __attribute__((ext_vector_type(16))) __bf16 v16bf;
typedef __attribute__((ext_vector_type(8)))  __bf16 v8bf;
typedef __attribute__((ext_vector_type(8)))  float  v8f;
typedef __attribute__((ext_vector_type(4)))  int    v4i;

#define LSEQ 1024
#define CDIM 1024
#define NHEAD 16
#define HDIM 64
#define WIN 64        // window_size // 2
#define NKT 10        // 10 key tiles of 16 -> 160-key padded window per wave
#define NKEY 160

__device__ __forceinline__ v8f wmma_bf16(v16bf a, v16bf b, v8f c) {
  return __builtin_amdgcn_wmma_f32_16x16x32_bf16(false, a, false, b, (short)0, c,
                                                 false, false);
}

// 16-byte global -> LDS copy. Async DMA (no VGPR round-trip) when available.
// Builtin signature (from compiler diagnostic): (v4i addrspace(1)*, v4i
// addrspace(3)*, i32 offset, i32 cpol).
__device__ __forceinline__ void g2l_b128(const __bf16* g, __bf16* l) {
#if __has_builtin(__builtin_amdgcn_global_load_async_to_lds_b128)
  v4i* gg = (v4i*)(void*)g;  // drop const, reinterpret as b128 in generic AS
  v4i* ll = (v4i*)(void*)l;
  __builtin_amdgcn_global_load_async_to_lds_b128(
      (__attribute__((address_space(1))) v4i*)gg,
      (__attribute__((address_space(3))) v4i*)ll, 0, 0);
#else
  *(v8bf*)l = *(const v8bf*)g;
#endif
}

__device__ __forceinline__ void wait_g2l() {
#if __has_builtin(__builtin_amdgcn_global_load_async_to_lds_b128)
#if __has_builtin(__builtin_amdgcn_s_wait_asynccnt)
  __builtin_amdgcn_s_wait_asynccnt(0);
#else
  asm volatile("s_wait_asynccnt 0x0" ::: "memory");
#endif
#endif
}

__global__ void cast_f32_bf16(const float* __restrict__ in,
                              __bf16* __restrict__ out, int n) {
  int i = blockIdx.x * blockDim.x + threadIdx.x;
  if (i < n) out[i] = (__bf16)in[i];
}

// ---------------------------------------------------------------------------
// D[M,N] = A[M,K] * W[N,K]^T + bias[N].  LDS-staged: block tile 128x256,
// K-step 32.  8 waves (2M x 4N), each wave computes a 64x64 tile (16 WMMAs).
// ---------------------------------------------------------------------------
#define BM 128
#define BN 256
#define BK 32

template <bool F32OUT>
__global__ void __launch_bounds__(256)
gemm_wmma(const __bf16* __restrict__ A, const __bf16* __restrict__ W,
          const float* __restrict__ bias, void* __restrict__ out,
          int M, int N, int K) {
  __shared__ __bf16 At[BM][BK];  //  8 KB, reused by 4 waves
  __shared__ __bf16 Wt[BN][BK];  // 16 KB, reused by 2 waves
  const int tid  = threadIdx.x;
  const int lane = tid & 31, wid = tid >> 5;
  const int half = lane >> 4, l16 = lane & 15;
  const int m0 = (wid & 1) * 64;   // wave tile origin inside the block
  const int n0 = (wid >> 1) * 64;
  const int gm = blockIdx.y * BM, gn = blockIdx.x * BN;

  v8f acc[4][4];
  v8f zf = {};
#pragma unroll
  for (int i = 0; i < 4; i++)
#pragma unroll
    for (int j = 0; j < 4; j++) acc[i][j] = zf;

  for (int k0 = 0; k0 < K; k0 += BK) {
    // ---- stage A (128x32) and W (256x32) tiles: 16B per chunk per thread ----
#pragma unroll
    for (int i = 0; i < 2; i++) {
      const int o = (tid + 256 * i) * 8;          // element offset in tile
      const int r = o >> 5, c = o & 31;
      g2l_b128(A + (size_t)(gm + r) * K + k0 + c, &At[r][c]);
    }
#pragma unroll
    for (int i = 0; i < 4; i++) {
      const int o = (tid + 256 * i) * 8;
      const int r = o >> 5, c = o & 31;
      g2l_b128(W + (size_t)(gn + r) * K + k0 + c, &Wt[r][c]);
    }
    wait_g2l();
    __syncthreads();

    // ---- fragments from LDS ----
    v16bf af[4], bw[4];
#pragma unroll
    for (int i = 0; i < 4; i++) {
      // A 16x32: lanes0-15 row=l16 K 0-15; lanes16-31 K 16-31
      const __bf16* p = &At[m0 + 16 * i + l16][half * 16];
#pragma unroll
      for (int e = 0; e < 8; e++) { af[i][e] = p[e]; af[i][8 + e] = p[8 + e]; }
    }
#pragma unroll
    for (int j = 0; j < 4; j++) {
      // B 32x16: lane column; lanes0-15 K 0-15, lanes16-31 K 16-31
      const __bf16* p = &Wt[n0 + 16 * j + l16][half * 16];
#pragma unroll
      for (int e = 0; e < 16; e++) bw[j][e] = p[e];
    }
#pragma unroll
    for (int i = 0; i < 4; i++)
#pragma unroll
      for (int j = 0; j < 4; j++)
        acc[i][j] = wmma_bf16(af[i], bw[j], acc[i][j]);
    __syncthreads();  // protect tiles before next stage overwrites them
  }

#pragma unroll
  for (int i = 0; i < 4; i++) {
#pragma unroll
    for (int j = 0; j < 4; j++) {
      const int col = gn + n0 + 16 * j + l16;
      const float bv = bias ? bias[col] : 0.0f;
#pragma unroll
      for (int r = 0; r < 8; r++) {
        const int row = gm + m0 + 16 * i + r + half * 8;  // C layout
        const float v = acc[i][j][r] + bv;
        if (F32OUT)
          ((float*)out)[(size_t)row * N + col] = v;
        else
          ((__bf16*)out)[(size_t)row * N + col] = (__bf16)v;
      }
    }
  }
}

// ---------------------------------------------------------------------------
// Banded attention: workgroup = (b, h, 128 q rows); wave = 16 q rows.
// The block's whole 256-key K/V window is staged in LDS once (async DMA),
// then S = QK^T, masked softmax, and O = PV run entirely out of LDS.
// ---------------------------------------------------------------------------
__global__ void __launch_bounds__(256)
attn_local(const __bf16* __restrict__ qkv, __bf16* __restrict__ attn_out) {
  __shared__ __bf16 Pt[8][16][NKEY];   // 40 KB probability staging (per wave)
  __shared__ __bf16 Kt[256][HDIM];     // 32 KB key window
  __shared__ __bf16 Vt[256][HDIM];     // 32 KB value window

  const int tid  = threadIdx.x;
  const int lane = tid & 31, wid = tid >> 5;
  const int half = lane >> 4, l16 = lane & 15;
  const int b = blockIdx.z, h = blockIdx.y;
  const int q0 = blockIdx.x * 128;
  const int qtile = q0 + wid * 16;
  const int kwin0 = q0 - WIN;          // block key window start (256 keys)
  const int kbase0 = qtile - WIN;      // wave key window start (160 padded)

  const size_t rs = 3 * CDIM;          // qkv row stride
  const __bf16* qb = qkv + (size_t)b * LSEQ * rs + h * HDIM;
  const __bf16* kb = qb + CDIM;

  // ---- stage K/V window: 256 keys x 64 dims, zero-fill OOB (NaN-safe) ----
#pragma unroll
  for (int i = 0; i < 8; i++) {
    const int o = (tid + 256 * i) * 8;
    const int r = o >> 6, c = o & 63;
    const int key = kwin0 + r;
    if (key >= 0 && key < LSEQ) {
      const __bf16* kp = kb + (size_t)key * rs + c;
      g2l_b128(kp, &Kt[r][c]);
      g2l_b128(kp + CDIM, &Vt[r][c]);  // V row sits CDIM after K row
    } else {
      v8bf z = {};
      *(v8bf*)&Kt[r][c] = z;
      *(v8bf*)&Vt[r][c] = z;
    }
  }

  // ---- Q fragments (two d-chunks of 32), direct from global ----
  v16bf aQ[2];
#pragma unroll
  for (int c = 0; c < 2; c++) {
    const __bf16* p = qb + (size_t)(qtile + l16) * rs + c * 32 + half * 16;
    v8bf lo = *(const v8bf*)p;
    v8bf hi = *(const v8bf*)(p + 8);
#pragma unroll
    for (int e = 0; e < 8; e++) { aQ[c][e] = lo[e]; aQ[c][8 + e] = hi[e]; }
  }

  wait_g2l();
  __syncthreads();

  // ---- S = Q K^T over 10 key tiles of 16 (from LDS) ----
  v8f S[NKT];
  v8f zf = {};
#pragma unroll
  for (int t = 0; t < NKT; t++) {
    S[t] = zf;
#pragma unroll
    for (int c = 0; c < 2; c++) {
      int lk = wid * 16 + 16 * t + l16;        // local key index
      lk = lk < 255 ? lk : 255;                // clamped reads are masked later
      const __bf16* p = &Kt[lk][c * 32 + half * 16];
      v16bf bK;
#pragma unroll
      for (int e = 0; e < 16; e++) bK[e] = p[e];
      S[t] = wmma_bf16(aQ[c], bK, S[t]);
    }
  }

  // ---- mask + softmax (rows live across 16 lanes of one half-wave) ----
  const float scale = 0.125f;  // 1/sqrt(64)
#pragma unroll
  for (int r = 0; r < 8; r++) {
    const int rowi = qtile + r + half * 8;
    float m = -3.0e38f;
#pragma unroll
    for (int t = 0; t < NKT; t++) {
      const int j = kbase0 + 16 * t + l16;
      const int d = rowi - j;
      float s = S[t][r] * scale;
      const bool ok = (j >= 0) && (j < LSEQ) && (d <= WIN) && (d >= -WIN);
      s = ok ? s : -3.0e38f;
      S[t][r] = s;
      m = fmaxf(m, s);
    }
    for (int msk = 1; msk < 16; msk <<= 1) m = fmaxf(m, __shfl_xor(m, msk, 32));
    float sum = 0.0f;
#pragma unroll
    for (int t = 0; t < NKT; t++) {
      const float e = __expf(S[t][r] - m);  // masked entries underflow to 0
      S[t][r] = e;
      sum += e;
    }
    for (int msk = 1; msk < 16; msk <<= 1) sum += __shfl_xor(sum, msk, 32);
    const float inv = 1.0f / sum;
#pragma unroll
    for (int t = 0; t < NKT; t++)
      Pt[wid][r + half * 8][16 * t + l16] = (__bf16)(S[t][r] * inv);
  }

  __syncthreads();  // uniform; orders P writes before A-layout reads

  // ---- O = P V: 160 keys in 5 chunks of 32; HD in 4 tiles of 16 ----
  v8f O[4];
#pragma unroll
  for (int jn = 0; jn < 4; jn++) O[jn] = zf;
#pragma unroll
  for (int c = 0; c < 5; c++) {
    v16bf aP;
    {
      const __bf16* p = &Pt[wid][l16][32 * c + half * 16];
#pragma unroll
      for (int e = 0; e < 8; e++) { aP[e] = p[e]; aP[8 + e] = p[8 + e]; }
    }
#pragma unroll
    for (int jn = 0; jn < 4; jn++) {
      const int d = jn * 16 + l16;
      v16bf bV;
#pragma unroll
      for (int e = 0; e < 16; e++) {
        int lk = wid * 16 + 32 * c + half * 16 + e;
        lk = lk < 255 ? lk : 255;            // P==0 there, value irrelevant
        bV[e] = Vt[lk][d];
      }
      O[jn] = wmma_bf16(aP, bV, O[jn]);
    }
  }

#pragma unroll
  for (int jn = 0; jn < 4; jn++) {
#pragma unroll
    for (int r = 0; r < 8; r++) {
      const int row = qtile + r + half * 8;
      attn_out[((size_t)(b * LSEQ + row)) * CDIM + h * HDIM + jn * 16 + l16] =
          (__bf16)O[jn][r];
    }
  }
}

extern "C" void kernel_launch(void* const* d_in, const int* in_sizes, int n_in,
                              void* d_out, int out_size, void* d_ws, size_t ws_size,
                              hipStream_t stream) {
  const float* x  = (const float*)d_in[0];   // [8,1024,1024]
  const float* wq = (const float*)d_in[1];   // [3072,1024]
  const float* bq = (const float*)d_in[2];   // [3072]
  const float* wo = (const float*)d_in[3];   // [1024,1024]
  const float* bo = (const float*)d_in[4];   // [1024]

  char* ws = (char*)d_ws;                    // needs 88 MB
  __bf16* xb   = (__bf16*)(ws);                          // 16 MB
  __bf16* wqb  = (__bf16*)(ws + (size_t)(16u << 20));    //  6 MB
  __bf16* wob  = (__bf16*)(ws + (size_t)(22u << 20));    //  2 MB
  __bf16* qkv  = (__bf16*)(ws + (size_t)(24u << 20));    // 48 MB
  __bf16* attn = (__bf16*)(ws + (size_t)(72u << 20));    // 16 MB

  const int NX = 8 * 1024 * 1024, NWQ = 3072 * 1024, NWO = 1024 * 1024;
  cast_f32_bf16<<<(NX + 255) / 256, 256, 0, stream>>>(x, xb, NX);
  cast_f32_bf16<<<(NWQ + 255) / 256, 256, 0, stream>>>(wq, wqb, NWQ);
  cast_f32_bf16<<<(NWO + 255) / 256, 256, 0, stream>>>(wo, wob, NWO);

  // QKV projection: [8192,1024] x [1024,3072] -> bf16 qkv
  dim3 g1(3072 / BN, 8192 / BM);
  gemm_wmma<false><<<g1, 256, 0, stream>>>(xb, wqb, bq, qkv, 8192, 3072, 1024);

  // Banded attention
  dim3 g2(LSEQ / 128, NHEAD, 8);
  attn_local<<<g2, 256, 0, stream>>>(qkv, attn);

  // Output projection: [8192,1024] x [1024,1024] -> f32 d_out
  dim3 g3(1024 / BN, 8192 / BM);
  gemm_wmma<true><<<g3, 256, 0, stream>>>(attn, wob, bo, d_out, 8192, 1024, 1024);
}
